// MSDA_4535485464952
// MI455X (gfx1250) — matrix operand order repaired and checked by reading, not compiled, not run on hardware
//
#include <hip/hip_runtime.h>

// ---------------------------------------------------------------------------
// Problem constants (fixed by the reference)
// ---------------------------------------------------------------------------
#define NC       6
#define NQ       40000
#define EMB      256
#define NH       8
#define HD       32
#define NL       4
#define ND       4
#define STOT     19560
#define MLEN     6000
#define ROWS     (NC * MLEN)        // 36000

typedef _Float16 half_t;
typedef __attribute__((ext_vector_type(16))) _Float16 v16h;
typedef __attribute__((ext_vector_type(8)))  _Float16 v8h;
typedef __attribute__((ext_vector_type(8)))  float    v8f;

// Level geometry
__device__ __constant__ int   c_LS[NL] = {0, 14720, 18400, 19320};
__device__ __constant__ int   c_H [NL] = {92, 46, 23, 12};
__device__ __constant__ int   c_W [NL] = {160, 80, 40, 20};

__device__ __forceinline__ void wait_asynccnt0() {
#if __has_builtin(__builtin_amdgcn_s_wait_asynccnt)
    __builtin_amdgcn_s_wait_asynccnt(0);
#else
    asm volatile("s_wait_asynccnt 0x0" ::: "memory");
#endif
}

// ---------------------------------------------------------------------------
// Weight prep: transpose + convert  Wt[n*K+k] = (f16) W[k*N+n]
// ---------------------------------------------------------------------------
__global__ void transpose_w_kernel(const float* __restrict__ W,
                                   half_t* __restrict__ Wt, int K, int N) {
    int t = blockIdx.x * 256 + threadIdx.x;
    if (t >= K * N) return;
    int n = t / K;
    int k = t - n * K;
    Wt[(size_t)n * K + k] = (half_t)W[(size_t)k * N + n];
}

// ---------------------------------------------------------------------------
// hit flags, per-query camera count, pos init
// ---------------------------------------------------------------------------
__global__ void hit_count_kernel(const int* __restrict__ bev_mask,
                                 int* __restrict__ hitmask,
                                 float* __restrict__ count_norm,
                                 int* __restrict__ pos) {
    int q = blockIdx.x * 256 + threadIdx.x;
    if (q >= NQ) return;
    int cnt = 0;
#pragma unroll
    for (int cam = 0; cam < NC; ++cam) {
        const int* bm = bev_mask + ((size_t)cam * NQ + q) * ND;
        int s = bm[0] + bm[1] + bm[2] + bm[3];
        int h = (s > 0) ? 1 : 0;
        hitmask[cam * NQ + q] = h;
        pos[cam * NQ + q] = -1;
        cnt += h;
    }
    count_norm[q] = 1.0f / (float)(cnt > 1 ? cnt : 1);
}

// ---------------------------------------------------------------------------
// Stable partition per camera: hit indices (ascending) first, then non-hit,
// truncated to MLEN. One block (1024 thr = 32 wave32) per camera.
// Deterministic (block-scan, no atomics).
// ---------------------------------------------------------------------------
__global__ void __launch_bounds__(1024)
partition_kernel(const int* __restrict__ hitmask,
                 int* __restrict__ idx,
                 int* __restrict__ validArr,
                 int* __restrict__ pos) {
    const int cam  = blockIdx.x;
    const int tid  = threadIdx.x;
    const int lane = tid & 31;
    const int wid  = tid >> 5;

    __shared__ int wsum[32];
    __shared__ int wpre[32];
    __shared__ int sbase;
    if (tid == 0) sbase = 0;
    __syncthreads();

    for (int pass = 0; pass < 2; ++pass) {
        for (int start = 0; start < NQ; start += 1024) {
            int q = start + tid;
            int flag = 0;
            if (q < NQ) {
                int h = hitmask[cam * NQ + q];
                flag = (pass == 0) ? h : (1 - h);
            }
            unsigned bal = (unsigned)__ballot(flag);
            int wprefix = __popc(bal & ((1u << lane) - 1u));
            if (lane == 0) wsum[wid] = __popc(bal);
            __syncthreads();
            if (tid == 0) {
                int run = sbase;
#pragma unroll
                for (int i = 0; i < 32; ++i) { wpre[i] = run; run += wsum[i]; }
            }
            __syncthreads();
            int offs = wpre[wid] + wprefix;
            if (flag && offs < MLEN) {
                idx[cam * MLEN + offs]      = q;
                validArr[cam * MLEN + offs] = (pass == 0) ? 1 : 0;
                if (pass == 0) pos[cam * NQ + q] = offs;
            }
            __syncthreads();
            if (tid == 0) sbase = wpre[31] + wsum[31];
            __syncthreads();
        }
    }
}

// ---------------------------------------------------------------------------
// Gather q_rb = query[idx]  (ROWS x 256 f32), float4 per thread
// ---------------------------------------------------------------------------
__global__ void gather_qrb_kernel(const float* __restrict__ query,
                                  const int* __restrict__ idx,
                                  float* __restrict__ q_rb) {
    long t = (long)blockIdx.x * 256 + threadIdx.x;   // ROWS*64 threads
    int row = (int)(t >> 6);
    int c4  = (int)(t & 63) * 4;
    if (row >= ROWS) return;
    int q = idx[row];
    float4 v = *(const float4*)(query + (size_t)q * EMB + c4);
    *(float4*)(q_rb + (size_t)row * EMB + c4) = v;
}

// ---------------------------------------------------------------------------
// WMMA GEMM:  C[M,N] = A[M,K](f32) * Wt[N,K](f16, pre-transposed) + bias
//             (+ residual)
// Block tile 64x64, K-step 32, 8 waves; each wave -> 16x32 via two
// v_wmma_f32_16x16x32_f16.
//   A: f32 global -> packed f16 cvt -> ds_store_b128 into LDS.
//   B: global_load_async_to_lds_b128 (CDNA5 async copy, ASYNCcnt-tracked),
//      one 16B transfer per thread per K-step, no VGPR round trip.
// ---------------------------------------------------------------------------
#define BM 64
#define BN 64
#define BK 32
#define ASTR 48   // halves per A row in LDS (16B-aligned, padded)
#define BSTR 48   // halves per B row (n-major) in LDS

template <bool HALF_OUT>
__global__ void __launch_bounds__(256)
gemm_wmma_kernel(const float* __restrict__ A, const half_t* __restrict__ Wt,
                 const float* __restrict__ bias, const float* __restrict__ resid,
                 void* __restrict__ Cout, int Mrows, int Ncols, int Kdim) {
    __shared__ half_t As[BM * ASTR];
    __shared__ half_t Bs[BN * BSTR];

    const int tid  = threadIdx.x;
    const int lane = tid & 31;
    const int wid  = tid >> 5;
    const int mbase = blockIdx.y * BM;
    const int nbase = blockIdx.x * BN;
    const int wr = wid & 3;     // 0..3 : M sub-tile
    const int wc = wid >> 2;    // 0..1 : N sub-tile (32 wide)

    v8f acc0 = {};
    v8f acc1 = {};

    // staging thread mapping
    const int arow = tid >> 2;           // 0..63
    const int acol = (tid & 3) * 8;      // 0,8,16,24
    const int brow = tid >> 2;           // 0..63  (n within tile)
    const int bkc  = (tid & 3) * 8;      // 0,8,16,24 (k chunk)

    int aSrcRow = mbase + arow;
    if (aSrcRow >= Mrows) aSrcRow = Mrows - 1;   // clamp (extra rows discarded)

    // constant per-thread addresses for B async staging
    const unsigned bLds = (unsigned)(uintptr_t)(&Bs[brow * BSTR + bkc]);
    const half_t* bGlb0 = Wt + (size_t)(nbase + brow) * Kdim + bkc;

    const int fm = lane & 15;
    const int kb = (lane >> 4) * 8;      // unused; A fragment also uses kh {0,16}
    const int kh = (lane >> 4) * 16;     // B fragment K-offsets {0,16}

    for (int k0 = 0; k0 < Kdim; k0 += BK) {
        // --- stage B: async global -> LDS, 16 bytes/thread ---
        {
            unsigned long long ga = (unsigned long long)(uintptr_t)(bGlb0 + k0);
            asm volatile("global_load_async_to_lds_b128 %0, %1, off"
                         :: "v"(bLds), "v"(ga) : "memory");
        }
        // --- stage A (f32 -> f16) ---
        {
            const float* ap = A + (size_t)aSrcRow * Kdim + k0 + acol;
            float4 f0 = *(const float4*)(ap);
            float4 f1 = *(const float4*)(ap + 4);
            v8h h;
            h[0] = (half_t)f0.x; h[1] = (half_t)f0.y;
            h[2] = (half_t)f0.z; h[3] = (half_t)f0.w;
            h[4] = (half_t)f1.x; h[5] = (half_t)f1.y;
            h[6] = (half_t)f1.z; h[7] = (half_t)f1.w;
            *(v8h*)(&As[arow * ASTR + acol]) = h;
        }
        wait_asynccnt0();
        __syncthreads();

        // --- fragments ---
        v8h a_lo = *(const v8h*)(&As[(wr * 16 + fm) * ASTR + kh]);
        v8h a_hi = *(const v8h*)(&As[(wr * 16 + fm) * ASTR + kh + 8]);
        v16h afrag = __builtin_shufflevector(a_lo, a_hi,
                        0,1,2,3,4,5,6,7,8,9,10,11,12,13,14,15);

        v8h b0l = *(const v8h*)(&Bs[(wc * 32 + fm) * BSTR + kh]);
        v8h b0h = *(const v8h*)(&Bs[(wc * 32 + fm) * BSTR + kh + 8]);
        v16h bfrag0 = __builtin_shufflevector(b0l, b0h,
                        0,1,2,3,4,5,6,7,8,9,10,11,12,13,14,15);

        v8h b1l = *(const v8h*)(&Bs[(wc * 32 + 16 + fm) * BSTR + kh]);
        v8h b1h = *(const v8h*)(&Bs[(wc * 32 + 16 + fm) * BSTR + kh + 8]);
        v16h bfrag1 = __builtin_shufflevector(b1l, b1h,
                        0,1,2,3,4,5,6,7,8,9,10,11,12,13,14,15);

        acc0 = __builtin_amdgcn_wmma_f32_16x16x32_f16(
                   false, afrag, false, bfrag0, (short)0, acc0, false, false);
        acc1 = __builtin_amdgcn_wmma_f32_16x16x32_f16(
                   false, afrag, false, bfrag1, (short)0, acc1, false, false);
        __syncthreads();
    }

    // --- epilogue: bias, residual, store ---
    const int mo    = (lane >> 4) * 8;
    const int ncol0 = nbase + wc * 32 + fm;
    const int ncol1 = ncol0 + 16;
    const float b0  = bias ? bias[ncol0] : 0.0f;
    const float b1  = bias ? bias[ncol1] : 0.0f;

#pragma unroll
    for (int r = 0; r < 8; ++r) {
        int mrow = mbase + wr * 16 + mo + r;
        if (mrow >= Mrows) continue;
        float v0 = acc0[r] + b0;
        float v1 = acc1[r] + b1;
        if (resid) {
            v0 += resid[(size_t)mrow * Ncols + ncol0];
            v1 += resid[(size_t)mrow * Ncols + ncol1];
        }
        if (HALF_OUT) {
            half_t* C = (half_t*)Cout;
            C[(size_t)mrow * Ncols + ncol0] = (half_t)v0;
            C[(size_t)mrow * Ncols + ncol1] = (half_t)v1;
        } else {
            float* C = (float*)Cout;
            C[(size_t)mrow * Ncols + ncol0] = v0;
            C[(size_t)mrow * Ncols + ncol1] = v1;
        }
    }
}

// ---------------------------------------------------------------------------
// Softmax over 16 attention logits per (row, head)
// ---------------------------------------------------------------------------
__global__ void softmax16_kernel(float* __restrict__ attn, int rows) {
    int r = blockIdx.x * 256 + threadIdx.x;
    if (r >= rows) return;
    float* p = attn + (size_t)r * 16;
    float mx = p[0];
#pragma unroll
    for (int i = 1; i < 16; ++i) mx = fmaxf(mx, p[i]);
    float s = 0.0f;
    float e[16];
#pragma unroll
    for (int i = 0; i < 16; ++i) { e[i] = __expf(p[i] - mx); s += e[i]; }
    float inv = 1.0f / s;
#pragma unroll
    for (int i = 0; i < 16; ++i) p[i] = e[i] * inv;
}

// ---------------------------------------------------------------------------
// Multi-scale deformable bilinear sampling.
// Block = one (cam, m) row; wave = head; lane = channel. Skips invalid rows
// (their output is never read by the combine kernel).
// ---------------------------------------------------------------------------
__global__ void __launch_bounds__(256)
sampler_kernel(const half_t* __restrict__ v,
               const float* __restrict__ offb,
               const float* __restrict__ attn,
               const float* __restrict__ refpts,
               const int* __restrict__ idx,
               const int* __restrict__ validArr,
               float* __restrict__ out_cam) {
    const int row = blockIdx.x;              // 0..ROWS-1
    if (!validArr[row]) return;
    const int cam  = row / MLEN;
    const int q    = idx[row];
    const int h    = threadIdx.x >> 5;       // head
    const int lane = threadIdx.x & 31;       // channel

    const float* offp  = offb + (size_t)row * EMB + h * (NL * ND * 2);
    const float* attnp = attn + (size_t)row * (NH * NL * ND) + h * (NL * ND);
    const float* refp  = refpts + ((size_t)cam * NQ + q) * (ND * 2);
    const half_t* vb   = v + (size_t)cam * STOT * EMB + h * HD + lane;

    float acc = 0.0f;
#pragma unroll
    for (int l = 0; l < NL; ++l) {
        const int   iH = c_H[l], iW = c_W[l], ls = c_LS[l];
        const float fH = (float)iH, fW = (float)iW;
#pragma unroll
        for (int d = 0; d < ND; ++d) {
            float rx = refp[d * 2 + 0];
            float ry = refp[d * 2 + 1];
            float ox = offp[l * (ND * 2) + d * 2 + 0];
            float oy = offp[l * (ND * 2) + d * 2 + 1];
            float x = rx * fW + ox - 0.5f;
            float y = ry * fH + oy - 0.5f;
            float x0f = floorf(x), y0f = floorf(y);
            float lx = x - x0f, ly = y - y0f;
            int x0 = (int)x0f, y0 = (int)y0f;
            float aw = attnp[l * ND + d];

#pragma unroll
            for (int cy = 0; cy < 2; ++cy) {
#pragma unroll
                for (int cx = 0; cx < 2; ++cx) {
                    int yi = y0 + cy, xi = x0 + cx;
                    bool ok = (yi >= 0) && (yi < iH) && (xi >= 0) && (xi < iW);
                    int yc = yi < 0 ? 0 : (yi > iH - 1 ? iH - 1 : yi);
                    int xc = xi < 0 ? 0 : (xi > iW - 1 ? iW - 1 : xi);
                    float wgt = (cx ? lx : 1.0f - lx) * (cy ? ly : 1.0f - ly);
                    int s = ls + yc * iW + xc;
                    float val = (float)vb[(size_t)s * EMB];
                    acc += ok ? aw * wgt * val : 0.0f;
                }
            }
        }
    }
    out_cam[(size_t)row * EMB + h * HD + lane] = acc;
}

// ---------------------------------------------------------------------------
// Deterministic combine: slots[q] = count_norm[q] * sum_cam out_cam[pos]
// ---------------------------------------------------------------------------
__global__ void combine_slots_kernel(const float* __restrict__ out_cam,
                                     const int* __restrict__ pos,
                                     const float* __restrict__ count_norm,
                                     float* __restrict__ slots) {
    long t = (long)blockIdx.x * 256 + threadIdx.x;   // NQ*64 threads
    int q  = (int)(t >> 6);
    int c4 = (int)(t & 63) * 4;
    if (q >= NQ) return;
    float4 acc = make_float4(0.f, 0.f, 0.f, 0.f);
#pragma unroll
    for (int cam = 0; cam < NC; ++cam) {
        int p = pos[cam * NQ + q];
        if (p >= 0) {
            float4 v = *(const float4*)(out_cam +
                        ((size_t)(cam * MLEN + p)) * EMB + c4);
            acc.x += v.x; acc.y += v.y; acc.z += v.z; acc.w += v.w;
        }
    }
    float cn = count_norm[q];
    acc.x *= cn; acc.y *= cn; acc.z *= cn; acc.w *= cn;
    *(float4*)(slots + (size_t)q * EMB + c4) = acc;
}

// ---------------------------------------------------------------------------
// Launcher
// ---------------------------------------------------------------------------
extern "C" void kernel_launch(void* const* d_in, const int* in_sizes, int n_in,
                              void* d_out, int out_size, void* d_ws, size_t ws_size,
                              hipStream_t stream) {
    const float* query  = (const float*)d_in[0];
    const float* refpts = (const float*)d_in[1];
    const int*   bev    = (const int*)  d_in[2];
    const float* value  = (const float*)d_in[3];
    // d_in[4] = max_len scalar (compile-time fixed to 6000)
    const float* W_off  = (const float*)d_in[5];
    const float* b_off  = (const float*)d_in[6];
    const float* W_attn = (const float*)d_in[7];
    const float* b_attn = (const float*)d_in[8];
    const float* W_val  = (const float*)d_in[9];
    const float* b_val  = (const float*)d_in[10];
    const float* W_out  = (const float*)d_in[11];
    const float* b_out  = (const float*)d_in[12];

    // workspace carve-out (256B aligned)
    char* ws = (char*)d_ws;
    size_t off = 0;
    auto carve = [&](size_t bytes) -> void* {
        void* p = (void*)(ws + off);
        off += (bytes + 255) & ~(size_t)255;
        return p;
    };
    half_t* Wv_t    = (half_t*)carve((size_t)EMB * EMB * sizeof(half_t));
    half_t* Woff_t  = (half_t*)carve((size_t)EMB * EMB * sizeof(half_t));
    half_t* Wattn_t = (half_t*)carve((size_t)EMB * 128 * sizeof(half_t));
    half_t* Wout_t  = (half_t*)carve((size_t)EMB * EMB * sizeof(half_t));
    int*    hitmask = (int*)carve((size_t)NC * NQ * sizeof(int));
    int*    posArr  = (int*)carve((size_t)NC * NQ * sizeof(int));
    int*    idxArr  = (int*)carve((size_t)ROWS * sizeof(int));
    int*    valArr  = (int*)carve((size_t)ROWS * sizeof(int));
    float*  cnorm   = (float*)carve((size_t)NQ * sizeof(float));
    float*  q_rb    = (float*)carve((size_t)ROWS * EMB * sizeof(float));
    half_t* v_h     = (half_t*)carve((size_t)NC * STOT * EMB * sizeof(half_t));
    float*  offbuf  = (float*)carve((size_t)ROWS * EMB * sizeof(float));
    float*  attnbuf = (float*)carve((size_t)ROWS * 128 * sizeof(float));
    float*  outcam  = (float*)carve((size_t)ROWS * EMB * sizeof(float));
    float*  slots   = (float*)carve((size_t)NQ * EMB * sizeof(float));
    (void)ws_size; (void)in_sizes; (void)n_in; (void)out_size;

    // 1) weights: transpose + convert to f16, Wt[N][K]
    transpose_w_kernel<<<(EMB*EMB + 255)/256, 256, 0, stream>>>(W_val,  Wv_t,    EMB, EMB);
    transpose_w_kernel<<<(EMB*EMB + 255)/256, 256, 0, stream>>>(W_off,  Woff_t,  EMB, EMB);
    transpose_w_kernel<<<(EMB*128 + 255)/256, 256, 0, stream>>>(W_attn, Wattn_t, EMB, 128);
    transpose_w_kernel<<<(EMB*EMB + 255)/256, 256, 0, stream>>>(W_out,  Wout_t,  EMB, EMB);

    // 2) hit flags + count_norm + pos init
    hit_count_kernel<<<(NQ + 255)/256, 256, 0, stream>>>(bev, hitmask, cnorm, posArr);

    // 3) stable partition per camera
    partition_kernel<<<NC, 1024, 0, stream>>>(hitmask, idxArr, valArr, posArr);

    // 4) gather q_rb
    gather_qrb_kernel<<<(ROWS*64 + 255)/256, 256, 0, stream>>>(query, idxArr, q_rb);

    // 5) value projection -> f16  (117360 x 256 @ 256x256)
    {
        const int Mv = NC * STOT;
        dim3 grid(EMB / BN, (Mv + BM - 1) / BM);
        gemm_wmma_kernel<true><<<grid, 256, 0, stream>>>(
            value, Wv_t, b_val, nullptr, (void*)v_h, Mv, EMB, EMB);
    }
    // 6) offset projection (36000 x 256 @ 256x256)
    {
        dim3 grid(EMB / BN, (ROWS + BM - 1) / BM);
        gemm_wmma_kernel<false><<<grid, 256, 0, stream>>>(
            q_rb, Woff_t, b_off, nullptr, (void*)offbuf, ROWS, EMB, EMB);
    }
    // 7) attention projection (36000 x 128 @ 256x128)
    {
        dim3 grid(128 / BN, (ROWS + BM - 1) / BM);
        gemm_wmma_kernel<false><<<grid, 256, 0, stream>>>(
            q_rb, Wattn_t, b_attn, nullptr, (void*)attnbuf, ROWS, 128, EMB);
    }
    // 8) softmax over 16 points per (row, head)
    softmax16_kernel<<<(ROWS*NH + 255)/256, 256, 0, stream>>>(attnbuf, ROWS*NH);

    // 9) deformable bilinear sampling
    sampler_kernel<<<ROWS, 256, 0, stream>>>(v_h, offbuf, attnbuf, refpts,
                                             idxArr, valArr, outcam);

    // 10) deterministic slot combine
    combine_slots_kernel<<<(NQ*64 + 255)/256, 256, 0, stream>>>(
        outcam, posArr, cnorm, slots);

    // 11) output projection + bias + residual -> d_out (40000 x 256)
    {
        dim3 grid(EMB / BN, (NQ + BM - 1) / BM);
        gemm_wmma_kernel<false><<<grid, 256, 0, stream>>>(
            slots, Wout_t, b_out, query, d_out, NQ, EMB, EMB);
    }
}
